// MambaBlock_65283502899305
// MI455X (gfx1250) — compile-verified, matrix operand order checked by reading, not run on hardware
//
#include <hip/hip_runtime.h>
#include <hip/hip_bf16.h>
#include <stdint.h>

// Mamba dims (fixed by reference)
#define B2   2
#define LSEQ 2048
#define DM   1024
#define DI   2048
#define NS   16
#define RR   64
#define KC   4
#define MT   (B2*LSEQ)   // 4096 rows for all GEMMs
#define XDW  128         // padded x_dbl row width (96 -> 128)

typedef __attribute__((ext_vector_type(16))) __bf16 v16bf;
typedef __attribute__((ext_vector_type(8)))  float  v8f;

__device__ __forceinline__ __bf16 f2bf(float f) {
  unsigned u = __builtin_bit_cast(unsigned, f);
  u += 0x7fffu + ((u >> 16) & 1u);          // round-to-nearest-even
  unsigned short h = (unsigned short)(u >> 16);
  return __builtin_bit_cast(__bf16, h);
}
__device__ __forceinline__ float fsigmoid(float x) { return 1.f / (1.f + __expf(-x)); }

// ---------------- f32 -> bf16 conversion ----------------
__global__ void k_cvt_bf16(const float* __restrict__ in, __bf16* __restrict__ out, int n) {
  int i = blockIdx.x * blockDim.x + threadIdx.x;
  if (i < n) out[i] = f2bf(in[i]);
}

// convert + zero-pad rows: out has padRows rows of `cols`, in has inRows rows
__global__ void k_cvt_pad_rows(const float* __restrict__ in, __bf16* __restrict__ out,
                               int inRows, int padRows, int cols) {
  int i = blockIdx.x * blockDim.x + threadIdx.x;
  if (i >= padRows * cols) return;
  int r = i / cols, c = i - r * cols;
  out[i] = (r < inRows) ? f2bf(in[(size_t)r * cols + c]) : f2bf(0.f);
}

// dt columns (first 64 of the 128-wide padded x_dbl rows) -> bf16, packed lda=64
__global__ void k_dt_extract(const float* __restrict__ xdbl, __bf16* __restrict__ out, int n) {
  int i = blockIdx.x * blockDim.x + threadIdx.x;
  if (i < n) out[i] = f2bf(xdbl[(size_t)(i >> 6) * XDW + (i & 63)]);
}

// ---------------- WMMA bf16 NT GEMM: C[M,N] = A[M,K] * B[N,K]^T ----------------
// All strides are compile-time: fragment loads are base + immediate offset.
// 256 threads = 8 waves; wave owns a 32x64 C tile (2 M-tiles x 4 N-tiles).
// K-loop software-pipelined (register double buffering), KK % 64 == 0.
// N must be a multiple of 64 (padded); M a multiple of 256.
// EP==0: plain f32 store.  EP==1: softplus(v + bias[col]).
template<int EP, int KK, int LDA, int LDB, int LDC>
__global__ void __launch_bounds__(256, 1)
k_gemm(const __bf16* __restrict__ A, const __bf16* __restrict__ B,
       float* __restrict__ C, const float* __restrict__ bias) {
  const int lane  = threadIdx.x & 31;
  const int wave  = threadIdx.x >> 5;
  const int l16   = lane & 15;
  const int khalf = (lane >> 4) * 16;          // lanes 16..31 hold K=16..31 of each 32-chunk

  const int mbase = blockIdx.y * 256 + wave * 32;
  const int nbase = blockIdx.x * 64;

  const __bf16* aP = A + (size_t)(mbase + l16) * LDA + khalf;  // + k0 / +16*LDA imm
  const __bf16* bP = B + (size_t)(nbase + l16) * LDB + khalf;  // + k0 + t*16*LDB imm

  v8f acc[8];
#pragma unroll
  for (int t = 0; t < 8; ++t)
#pragma unroll
    for (int i = 0; i < 8; ++i) acc[t][i] = 0.f;

  v16bf a0[2], a1[2], b0[4], b1[4];
  a0[0] = *reinterpret_cast<const v16bf*>(aP);
  a0[1] = *reinterpret_cast<const v16bf*>(aP + 16 * LDA);
#pragma unroll
  for (int t = 0; t < 4; ++t)
    b0[t] = *reinterpret_cast<const v16bf*>(bP + t * 16 * LDB);

  for (int k0 = 0; k0 < KK; k0 += 64) {
    // stream-prefetch A two chunks ahead (global_prefetch_b8); B stays L2-resident
    if (k0 + 128 < KK) {
      __builtin_prefetch(aP + k0 + 128, 0, 1);
      __builtin_prefetch(aP + 16 * LDA + k0 + 128, 0, 1);
    }
    // load chunk k0+32 into buffer 1 (always valid: KK % 64 == 0)
    a1[0] = *reinterpret_cast<const v16bf*>(aP + k0 + 32);
    a1[1] = *reinterpret_cast<const v16bf*>(aP + 16 * LDA + k0 + 32);
#pragma unroll
    for (int t = 0; t < 4; ++t)
      b1[t] = *reinterpret_cast<const v16bf*>(bP + t * 16 * LDB + k0 + 32);

    // compute on buffer 0 (chunk k0) while buffer-1 loads are in flight
#pragma unroll
    for (int t = 0; t < 4; ++t) {
      acc[t]     = __builtin_amdgcn_wmma_f32_16x16x32_bf16(false, a0[0], false, b0[t],
                                                           (short)0, acc[t],     false, false);
      acc[4 + t] = __builtin_amdgcn_wmma_f32_16x16x32_bf16(false, a0[1], false, b0[t],
                                                           (short)0, acc[4 + t], false, false);
    }

    // load chunk k0+64 into buffer 0 (if any)
    if (k0 + 64 < KK) {
      a0[0] = *reinterpret_cast<const v16bf*>(aP + k0 + 64);
      a0[1] = *reinterpret_cast<const v16bf*>(aP + 16 * LDA + k0 + 64);
#pragma unroll
      for (int t = 0; t < 4; ++t)
        b0[t] = *reinterpret_cast<const v16bf*>(bP + t * 16 * LDB + k0 + 64);
    }

    // compute on buffer 1 (chunk k0+32)
#pragma unroll
    for (int t = 0; t < 4; ++t) {
      acc[t]     = __builtin_amdgcn_wmma_f32_16x16x32_bf16(false, a1[0], false, b1[t],
                                                           (short)0, acc[t],     false, false);
      acc[4 + t] = __builtin_amdgcn_wmma_f32_16x16x32_bf16(false, a1[1], false, b1[t],
                                                           (short)0, acc[4 + t], false, false);
    }
  }

  // epilogue: ISA C/D layout — VGPR i holds row (i + (lane>=16 ? 8:0)), col lane&15
  const int rowoff = mbase + ((lane >> 4) ? 8 : 0);
#pragma unroll
  for (int t = 0; t < 4; ++t) {
    const int col = nbase + t * 16 + l16;
    float* cP = C + (size_t)rowoff * LDC + col;
    float bv = (EP == 1) ? bias[col] : 0.f;
#pragma unroll
    for (int i = 0; i < 8; ++i) {
      float v0 = acc[t][i];
      float v1 = acc[4 + t][i];
      if (EP == 1) {
        v0 += bv; v0 = (v0 > 20.f) ? v0 : __logf(1.f + __expf(v0));
        v1 += bv; v1 = (v1 > 20.f) ? v1 : __logf(1.f + __expf(v1));
      }
      cP[(size_t)i * LDC]        = v0;
      cP[(size_t)(16 + i) * LDC] = v1;
    }
  }
}

// ---------------- causal depthwise conv (K=4) + SiLU; emits f32 + bf16 ----------------
__global__ void k_conv_silu(const float* __restrict__ xz, const float* __restrict__ cw,
                            const float* __restrict__ cb,
                            float* __restrict__ xh32, __bf16* __restrict__ xh16) {
  int tid = blockIdx.x * blockDim.x + threadIdx.x;
  if (tid >= MT * DI) return;
  int d = tid & (DI - 1);
  int m = tid >> 11;          // DI == 2048
  int l = m & (LSEQ - 1);
  int b = m >> 11;            // LSEQ == 2048
  float acc = cb[d];
#pragma unroll
  for (int k = 0; k < KC; ++k) {
    int li = l - (KC - 1) + k;
    if (li >= 0)
      acc += xz[(size_t)(b * LSEQ + li) * (2 * DI) + d] * cw[d * KC + k];
  }
  float v = acc * fsigmoid(acc);   // SiLU
  xh32[tid] = v;
  xh16[tid] = f2bf(v);
}

// ---------------- selective scan + D-skip + SiLU gate ----------------
// 16 lanes per (b,d) channel: lane n owns state n. Serial chain per step is a
// single exp+fma; y = sum_n h_n*C_n via 4-step shfl_xor tree within the 16-lane
// group. 65536 threads total (2048 wave32 waves).
__global__ void k_scan_gate(const float* __restrict__ delta, const float* __restrict__ xh32,
                            const float* __restrict__ xdbl,  const float* __restrict__ xz,
                            const float* __restrict__ A_log, const float* __restrict__ Dp,
                            __bf16* __restrict__ g16) {
  int tid = blockIdx.x * blockDim.x + threadIdx.x;   // 0..65535
  if (tid >= B2 * DI * NS) return;
  int n = tid & (NS - 1);
  int g = tid >> 4;            // channel id 0..4095
  int d = g & (DI - 1);
  int b = g >> 11;

  const float Areg = -__expf(A_log[(size_t)d * NS + n]);
  const float Dval = Dp[d];
  float h = 0.f;

  size_t rowD = (size_t)(b * LSEQ) * DI + d;            // delta / xh / g index
  size_t rowX = (size_t)(b * LSEQ) * XDW;               // padded x_dbl row base
  size_t rowZ = (size_t)(b * LSEQ) * (2 * DI) + DI + d; // residual (gate) index

  for (int l = 0; l < LSEQ; ++l) {
    float dlt = delta[rowD];                 // uniform across the 16-lane group
    float xv  = xh32[rowD];
    float Bv  = xdbl[rowX + RR + n];         // per-lane state inputs
    float Cv  = xdbl[rowX + RR + NS + n];

    h = __expf(dlt * Areg) * h + (dlt * xv) * Bv;   // ZOH A, Euler B
    float p = h * Cv;
    p += __shfl_xor(p, 1, 16);               // reduce over the 16 states
    p += __shfl_xor(p, 2, 16);
    p += __shfl_xor(p, 4, 16);
    p += __shfl_xor(p, 8, 16);

    if (n == 0) {
      float res = xz[rowZ];
      float yt  = p + xv * Dval;             // D skip
      g16[rowD] = f2bf(yt * res * fsigmoid(res));   // * SiLU(res)
    }
    rowD += DI; rowX += XDW; rowZ += 2 * DI;
  }
}

// ---------------- launcher ----------------
extern "C" void kernel_launch(void* const* d_in, const int* in_sizes, int n_in,
                              void* d_out, int out_size, void* d_ws, size_t ws_size,
                              hipStream_t stream) {
  const float* x        = (const float*)d_in[0];
  const float* in_w     = (const float*)d_in[1];
  const float* conv_w   = (const float*)d_in[2];
  const float* conv_b   = (const float*)d_in[3];
  const float* xp_w     = (const float*)d_in[4];
  const float* dt_w     = (const float*)d_in[5];
  const float* dt_b     = (const float*)d_in[6];
  const float* A_log    = (const float*)d_in[7];
  const float* Dp       = (const float*)d_in[8];
  const float* out_w    = (const float*)d_in[9];
  float* out = (float*)d_out;

  // workspace carve-up (all offsets 256B-aligned)
  char* p = (char*)d_ws;
  __bf16* x_bf   = (__bf16*)p;            p += (size_t)MT * DM * 2;        // 8 MB
  __bf16* win_bf = (__bf16*)p;            p += (size_t)2 * DI * DM * 2;    // 8 MB
  __bf16* wxp_bf = (__bf16*)p;            p += (size_t)XDW * DI * 2;       // .5 MB (padded 128 rows)
  __bf16* wdt_bf = (__bf16*)p;            p += (size_t)DI * RR * 2;        // .25 MB
  __bf16* wout_bf= (__bf16*)p;            p += (size_t)DM * DI * 2;        // 4 MB
  float*  xz     = (float*)p;             p += (size_t)MT * 2 * DI * 4;    // 64 MB
  float*  xh32   = (float*)p;             p += (size_t)MT * DI * 4;        // 32 MB
  __bf16* xh16   = (__bf16*)p;            p += (size_t)MT * DI * 2;        // 16 MB
  float*  xdbl   = (float*)p;             p += (size_t)MT * XDW * 4;       // 2 MB (padded)
  __bf16* dt_bf  = (__bf16*)p;            p += (size_t)MT * RR * 2;        // .5 MB
  float*  delta  = (float*)p;             p += (size_t)MT * DI * 4;        // 32 MB
  __bf16* g_bf   = (__bf16*)p;            p += (size_t)MT * DI * 2;        // 16 MB
  (void)ws_size;

  const int T = 256;
  // bf16 conversions
  { int n = MT * DM;      k_cvt_bf16<<<(n + T - 1) / T, T, 0, stream>>>(x,     x_bf,    n); }
  { int n = 2 * DI * DM;  k_cvt_bf16<<<(n + T - 1) / T, T, 0, stream>>>(in_w,  win_bf,  n); }
  { int n = XDW * DI;     k_cvt_pad_rows<<<(n + T - 1) / T, T, 0, stream>>>(xp_w, wxp_bf, 96, XDW, DI); }
  { int n = DI * RR;      k_cvt_bf16<<<(n + T - 1) / T, T, 0, stream>>>(dt_w,  wdt_bf,  n); }
  { int n = DM * DI;      k_cvt_bf16<<<(n + T - 1) / T, T, 0, stream>>>(out_w, wout_bf, n); }

  // GEMM1: xz = x @ in_proj_w.T   (M=4096, N=4096, K=1024)
  k_gemm<0, DM, DM, DM, 2 * DI><<<dim3((2 * DI) / 64, MT / 256), T, 0, stream>>>(
      x_bf, win_bf, xz, nullptr);

  // conv(K=4, causal) + SiLU  -> xh32, xh16
  { int n = MT * DI; k_conv_silu<<<(n + T - 1) / T, T, 0, stream>>>(xz, conv_w, conv_b, xh32, xh16); }

  // GEMM2: x_dbl = xh @ x_proj_w.T  (M=4096, N=128 padded, K=2048)
  k_gemm<0, DI, DI, DI, XDW><<<dim3(XDW / 64, MT / 256), T, 0, stream>>>(
      xh16, wxp_bf, xdbl, nullptr);

  // pack dt columns to bf16
  { int n = MT * RR; k_dt_extract<<<(n + T - 1) / T, T, 0, stream>>>(xdbl, dt_bf, n); }

  // GEMM3: delta = softplus(dt @ dt_proj_w.T + b)  (M=4096, N=2048, K=64)
  k_gemm<1, RR, RR, RR, DI><<<dim3(DI / 64, MT / 256), T, 0, stream>>>(
      dt_bf, wdt_bf, delta, dt_b);

  // selective scan + D-skip + SiLU gate -> g (bf16)
  { int n = B2 * DI * NS; k_scan_gate<<<(n + T - 1) / T, T, 0, stream>>>(
        delta, xh32, xdbl, xz, A_log, Dp, g_bf); }

  // GEMM4: out = g @ out_proj_w.T  (M=4096, N=1024, K=2048)
  k_gemm<0, DI, DI, DI, DM><<<dim3(DM / 64, MT / 256), T, 0, stream>>>(
      g_bf, wout_bf, out, nullptr);
}
